// A100OptimizedSparseSimilarity_9096740733739
// MI455X (gfx1250) — compile-verified, matrix operand order checked by reading, not run on hardware
//
#include <hip/hip_runtime.h>

// ---------------------------------------------------------------------------
// Cosine-similarity KNN (top-10 + softmax) -> COO triplets, for MI455X/gfx1250
//   fx: [49152,128] f32, fy: [50000,128] f32
//   out: [NX*K] f32 softmax values | [NX*K] i32 rows | [NX*K] i32 cols
// Strategy: normalize->bf16 once; similarity GEMM on v_wmma_f32_16x16x32_bf16;
// B tiles double-buffered in LDS via GLOBAL_LOAD_ASYNC_TO_LDS_B128 (ASYNCcnt);
// fused per-row top-10 maintained in registers.
// ---------------------------------------------------------------------------

#define NXv   49152
#define NYv   50000
#define NYP   50016      // NY padded to a multiple of 32 columns
#define Cdim  128
#define KTOP  10
#define TAUf  0.05f

typedef __attribute__((ext_vector_type(16))) __bf16 v16bf;
typedef __attribute__((ext_vector_type(8)))  __bf16 v8bf;
typedef __attribute__((ext_vector_type(8)))  float  v8f;

union V16U { v16bf v; v8bf h[2]; };

__device__ __forceinline__ unsigned short f2bf(float f) {
  unsigned u = __float_as_uint(f);
  u += 0x7FFFu + ((u >> 16) & 1u);          // round-to-nearest-even
  return (unsigned short)(u >> 16);
}

// --------------------------- phase 1: normalize ----------------------------
// One wave32 per vector of 128 floats; 8 vectors per 256-thread block.
__global__ __launch_bounds__(256) void norm_bf16_kernel(
    const float* __restrict__ in, unsigned short* __restrict__ out,
    int nvec, int nvecOut) {
  const int lane = threadIdx.x & 31;
  const int vec  = blockIdx.x * 8 + (threadIdx.x >> 5);
  if (vec >= nvecOut) return;

  float4 x = make_float4(0.f, 0.f, 0.f, 0.f);
  if (vec < nvec)
    x = *(const float4*)(in + (size_t)vec * Cdim + lane * 4);

  float ss = x.x * x.x + x.y * x.y + x.z * x.z + x.w * x.w;
  #pragma unroll
  for (int off = 16; off >= 1; off >>= 1)
    ss += __shfl_xor(ss, off, 32);

  const float inv = 1.0f / fmaxf(sqrtf(ss), 1e-12f);

  ushort4 o;
  o.x = f2bf(x.x * inv);
  o.y = f2bf(x.y * inv);
  o.z = f2bf(x.z * inv);
  o.w = f2bf(x.w * inv);
  *(ushort4*)(out + (size_t)vec * Cdim + lane * 4) = o;
}

// ------------------- phase 2: WMMA GEMM + fused top-k ----------------------
// Block = 256 threads = 8 waves; wave owns 16 rows; loop N in tiles of 32.
// B tiles are double-buffered in LDS via async global->LDS DMA loads.
__global__ __launch_bounds__(256) void knn_topk_kernel(
    const unsigned short* __restrict__ fxb,   // [NXv][128] bf16, row-major
    const unsigned short* __restrict__ fyb,   // [NYP][128] bf16, row-major
    float* __restrict__ out_vals,
    int*   __restrict__ out_rows,
    int*   __restrict__ out_cols) {
  __shared__ __align__(16) unsigned short ldsB[2][32 * 136]; // 2x (32 cols x 128 K), padded
  __shared__ __align__(16) float ldsSim[8][16 * 33];         // per-wave 16x32 tile, padded
  __shared__ float ldsMV[8][16][2][KTOP];
  __shared__ int   ldsMI[8][16][2][KTOP];

  const int tid   = threadIdx.x;
  const int wave  = tid >> 5;
  const int lane  = tid & 31;
  const int lrow  = lane & 15;
  const int lhalf = lane >> 4;
  const int h8    = lhalf * 8;               // K sub-offset per lane-half (A/B layout)
  const int myRow = blockIdx.x * 128 + wave * 16 + lrow;

  const int colB = tid >> 3;                 // 0..31: which column this thread stages
  const int koff = (tid & 7) * 16;           // 0..112: K offset (16 bf16 per thread)

  // Async global->LDS issue for one 32x128 bf16 B tile (32B per thread).
  auto issue_tile = [&](int buf, int nbase) {
    const unsigned short* g = fyb + (size_t)(nbase + colB) * Cdim + koff;
    unsigned lds = (unsigned)(uintptr_t)(&ldsB[buf][colB * 136 + koff]);
    unsigned long long ga = (unsigned long long)(uintptr_t)g;
    asm volatile("global_load_async_to_lds_b128 %0, %1, off"
                 :: "v"(lds), "v"(ga) : "memory");
    asm volatile("global_load_async_to_lds_b128 %0, %1, off offset:16"
                 :: "v"(lds), "v"(ga) : "memory");
  };

  // A fragments: 16 rows x 128 K, four 16x32 bf16 fragments, loaded once.
  v16bf afrag[4];
  #pragma unroll
  for (int kc = 0; kc < 4; ++kc) {
    const unsigned short* p = fxb + (size_t)myRow * Cdim + kc * 32 + h8;
    V16U t;
    t.h[0] = *(const v8bf*)(p);              // K = kc*32 + h8 .. +7
    t.h[1] = *(const v8bf*)(p + 16);         // K = kc*32 + 16 + h8 .. +7
    afrag[kc] = t.v;
  }

  // Per-lane sorted (descending) top-10 for (row=lrow, column-half=lhalf).
  float tv[KTOP]; int ti[KTOP];
  #pragma unroll
  for (int i = 0; i < KTOP; ++i) { tv[i] = -1e30f; ti[i] = 0; }

  issue_tile(0, 0);                          // prologue: stage tile 0

  int cur = 0;
  for (int nb = 0; nb < NYP; nb += 32) {
    if (nb + 32 < NYP) {
      issue_tile(cur ^ 1, nb + 32);          // overlap next tile with compute
      if (nb + 64 < NYP)                     // warm L2 one tile further ahead
        __builtin_prefetch(fyb + (size_t)(nb + 64 + colB) * Cdim + koff, 0, 1);
      asm volatile("s_wait_asynccnt 0x2" ::: "memory");  // older 2 loads landed
    } else {
      asm volatile("s_wait_asynccnt 0x0" ::: "memory");
    }
    __syncthreads();                         // tile `nb` visible to all waves

    // Preload all 8 B fragments, then issue the 8 WMMAs back-to-back.
    const unsigned short* bbase = &ldsB[cur][0];
    V16U bf[8];
    #pragma unroll
    for (int kc = 0; kc < 4; ++kc) {
      const unsigned short* b0p = bbase + lrow * 136 + kc * 32 + h8;
      const unsigned short* b1p = bbase + (16 + lrow) * 136 + kc * 32 + h8;
      bf[kc].h[0]     = *(const v8bf*)(b0p);
      bf[kc].h[1]     = *(const v8bf*)(b0p + 16);
      bf[kc + 4].h[0] = *(const v8bf*)(b1p);
      bf[kc + 4].h[1] = *(const v8bf*)(b1p + 16);
    }
    v8f acc0 = {}; v8f acc1 = {};
    #pragma unroll
    for (int kc = 0; kc < 4; ++kc) {
      acc0 = __builtin_amdgcn_wmma_f32_16x16x32_bf16(
          false, afrag[kc], false, bf[kc].v,     (short)0, acc0, false, false);
      acc1 = __builtin_amdgcn_wmma_f32_16x16x32_bf16(
          false, afrag[kc], false, bf[kc + 4].v, (short)0, acc1, false, false);
    }

    // Stage 16x32 f32 similarity tile to per-wave LDS (C layout: reg r, lane-half).
    float* sim = ldsSim[wave];
    #pragma unroll
    for (int r = 0; r < 8; ++r) {
      const int row = r + h8;
      sim[row * 33 + lrow]      = acc0[r];
      sim[row * 33 + 16 + lrow] = acc1[r];
    }
    __syncthreads();   // sim visible cross-lane; also: all WMMA reads of
                       // ldsB[cur] done before next iteration's async writes

    // Top-k scan: each lane scans its 16 candidates of its row / column-half.
    const float* srow  = &ldsSim[wave][lrow * 33 + lhalf * 16];
    const int    cbase = nb + lhalf * 16;
    #pragma unroll
    for (int j = 0; j < 16; ++j) {
      const float v = srow[j];
      const int   c = cbase + j;
      if ((c < NYv) && (v > tv[KTOP - 1])) {
        tv[KTOP - 1] = v; ti[KTOP - 1] = c;
        #pragma unroll
        for (int i = KTOP - 1; i > 0; --i) {
          if (tv[i] > tv[i - 1]) {
            float tf = tv[i]; tv[i] = tv[i - 1]; tv[i - 1] = tf;
            int   tc = ti[i]; ti[i] = ti[i - 1]; ti[i - 1] = tc;
          }
        }
      }
    }
    cur ^= 1;
  }

  // Merge the two column-half lists per row, softmax, and emit COO.
  __syncthreads();
  #pragma unroll
  for (int i = 0; i < KTOP; ++i) {
    ldsMV[wave][lrow][lhalf][i] = tv[i];
    ldsMI[wave][lrow][lhalf][i] = ti[i];
  }
  __syncthreads();

  if (lhalf == 0) {
    const float* va = ldsMV[wave][lrow][0];
    const float* vb = ldsMV[wave][lrow][1];
    const int*   ia = ldsMI[wave][lrow][0];
    const int*   ib = ldsMI[wave][lrow][1];
    float mv[KTOP]; int mi[KTOP];
    int a = 0, b = 0;
    #pragma unroll
    for (int i = 0; i < KTOP; ++i) {       // two-pointer merge, a+b==i<=9: in range
      const float fa = va[a], fb = vb[b];
      if (fa >= fb) { mv[i] = fa; mi[i] = ia[a]; ++a; }
      else          { mv[i] = fb; mi[i] = ib[b]; ++b; }
    }
    const float invTau = 1.0f / TAUf;
    float e[KTOP]; float s = 0.f;
    #pragma unroll
    for (int i = 0; i < KTOP; ++i) {       // mv[0] is the row max (sorted desc)
      e[i] = __expf((mv[i] - mv[0]) * invTau);
      s += e[i];
    }
    const float rs = 1.0f / s;
    const size_t base = (size_t)myRow * KTOP;
    #pragma unroll
    for (int i = 0; i < KTOP; ++i) {
      out_vals[base + i] = e[i] * rs;
      out_rows[base + i] = myRow;
      out_cols[base + i] = mi[i];
    }
  }
}

// ------------------------------- launcher ----------------------------------
extern "C" void kernel_launch(void* const* d_in, const int* in_sizes, int n_in,
                              void* d_out, int out_size, void* d_ws, size_t ws_size,
                              hipStream_t stream) {
  (void)in_sizes; (void)n_in; (void)out_size; (void)ws_size;
  const float* fx = (const float*)d_in[0];   // [1, NX, 128] f32
  const float* fy = (const float*)d_in[1];   // [1, NY, 128] f32

  unsigned short* fxb = (unsigned short*)d_ws;                 // NXv*128 bf16
  unsigned short* fyb = fxb + (size_t)NXv * Cdim;              // NYP*128 bf16

  float* out_vals = (float*)d_out;
  int*   out_rows = (int*)d_out + (size_t)NXv * KTOP;
  int*   out_cols = (int*)d_out + (size_t)2 * NXv * KTOP;

  norm_bf16_kernel<<<NXv / 8, 256, 0, stream>>>(fx, fxb, NXv, NXv);
  norm_bf16_kernel<<<NYP / 8, 256, 0, stream>>>(fy, fyb, NYv, NYP);
  knn_topk_kernel<<<NXv / 128, 256, 0, stream>>>(fxb, fyb, out_vals, out_rows, out_cols);
}
